// OCST_net_tcn_64287070487002
// MI455X (gfx1250) — compile-verified, matrix operand order checked by reading, not run on hardware
//
#include <hip/hip_runtime.h>

// ---------------------------------------------------------------------------
// OCST-net TCN forward for MI455X (gfx1250, wave32, WMMA).
// Heavy contractions (node diffusion x·A with K=512, end1 MLP K=416) run as
// f16-input / f32-accumulate v_wmma_f32_16x16x32_f16 tiled GEMMs.
// Small C=32 channel mixes / BN / softmax stay on VALU (memory-bound).
// ---------------------------------------------------------------------------

typedef __attribute__((ext_vector_type(16))) _Float16 v16h;
typedef __attribute__((ext_vector_type(8)))  _Float16 v8h;
typedef __attribute__((ext_vector_type(8)))  float    v8f;

// model constants
constexpr int kB = 64, kN = 512, kT = 12, kC = 32, kSK = 8, kEC = 512;
constexpr int kOD = 12, kDIMS = 10, kL = 8, kRF = 13;
constexpr int kSKCH = 416;  // SK * 52

// ---------------------------------------------------------------------------
// workspace arena (bytes, 256-aligned)
// ---------------------------------------------------------------------------
constexpr size_t AL(size_t x) { return (x + 255) & ~(size_t)255; }
constexpr size_t SZ_ACT32 = (size_t)kB * kRF * kC * kN * 4;   // 54.5 MB
constexpr size_t SZ_SKIP  = (size_t)kB * kSKCH * kN * 4;      // 54.5 MB
constexpr size_t SZ_ADJH  = AL((size_t)kN * kN * 2);
constexpr size_t SZ_A2PH  = (size_t)kB * kN * kN * 2;         // 33.5 MB
constexpr size_t SZ_ACT16 = (size_t)kB * kT * kC * kN * 2;    // 25.2 MB
constexpr size_t SZ_G32   = (size_t)kB * kT * kC * kN * 4;    // 50.3 MB

constexpr size_t O_X     = 0;
constexpr size_t O_XA    = O_X + SZ_ACT32;
constexpr size_t O_SKIP  = O_XA + SZ_ACT32;
constexpr size_t O_ADPH  = O_SKIP + SZ_SKIP;
constexpr size_t O_ADPAH = O_ADPH + SZ_ADJH;
constexpr size_t O_A2PH  = O_ADPAH + SZ_ADJH;
constexpr size_t O_T2    = O_A2PH + SZ_A2PH;
constexpr size_t O_PART  = O_T2 + AL((size_t)kB * kDIMS * kDIMS * 4);
constexpr size_t O_FSTAT = O_PART + AL((size_t)2 * 4096 * 4);
constexpr size_t O_E1WH  = O_FSTAT + AL(128 * 4);
constexpr size_t O_XGH   = O_E1WH + AL((size_t)kEC * kSKCH * 2);
constexpr size_t O_XAGH  = O_XGH + SZ_ACT16;
constexpr size_t O_T1H   = O_XAGH + SZ_ACT16;
constexpr size_t O_T2H   = O_T1H + SZ_ACT16;
constexpr size_t O_G     = O_T2H + SZ_ACT16;
constexpr size_t O_GA    = O_G + SZ_G32;
constexpr size_t WS_NEED = O_GA + SZ_G32;                     // ~400 MB
// end-stage aliases over dead per-block arena
constexpr size_t O_SKIPH = O_XGH;  // 27.3 MB over XGH+XAGH (50.3 MB)
constexpr size_t O_HH    = O_G;    // 33.5 MB over G (50.3 MB)

// ---------------------------------------------------------------------------
// WMMA tiled GEMM:  C[M,N] = A[M,K] * B[K,N], A/B f16 row-major, C f16,
// f32 accumulation. Block tile 128x64, 8 waves (wave32) arranged 4x2; each
// wave owns a 32x32 output patch (2x2 accumulators -> 4 WMMA per K-chunk,
// 8 ds_load_b128 per 4 WMMA). K must be a multiple of 32, N of 64.
// Optional batching via element strides (stride 0 broadcasts an operand).
// ---------------------------------------------------------------------------
__launch_bounds__(256)
__global__ void k_gemm_f16(const _Float16* __restrict__ A,
                           const _Float16* __restrict__ B,
                           _Float16* __restrict__ C,
                           int M, int Ntot, int K,
                           long sA, long sB, long sC) {
  __shared__ __align__(16) _Float16 sAt[128 * 32];  // [row][k]
  __shared__ __align__(16) _Float16 sBt[64 * 32];   // transposed: [col][k]
  const int tid  = threadIdx.x;
  const int lane = tid & 31;
  const int wid  = tid >> 5;
  const int wm   = wid & 3;   // 4 row-groups of 32
  const int wn   = wid >> 2;  // 2 col-groups of 32
  const long z   = blockIdx.z;
  const _Float16* Ab = A + z * sA;
  const _Float16* Bb = B + z * sB;
  _Float16*       Cb = C + z * sC;
  const int m0 = blockIdx.x * 128;
  const int n0 = blockIdx.y * 64;

  v8f acc00 = {}, acc01 = {}, acc10 = {}, acc11 = {};

  const int arow = tid >> 2;        // 0..63 (two passes cover 128 rows)
  const int akp  = (tid & 3) * 8;   // 0,8,16,24
  const int brow = tid >> 3;        // k 0..31
  const int bnp  = (tid & 7) * 8;   // 0..56

  const int hl = lane & 15;
  const int kb = (lane < 16) ? 0 : 8;  // CDNA5 16-bit A-frag K mapping

  for (int k0 = 0; k0 < K; k0 += 32) {
    // stage A tile (128x32, two passes; zero-fill out-of-range rows)
#pragma unroll
    for (int p = 0; p < 2; ++p) {
      v8h av = {};
      int gm = m0 + p * 64 + arow;
      if (gm < M) av = *(const v8h*)(Ab + (size_t)gm * K + k0 + akp);
      *(v8h*)(&sAt[(p * 64 + arow) * 32 + akp]) = av;
    }
    // stage B tile transposed ([col][k]) from row-major global
    v8h bv = *(const v8h*)(Bb + (size_t)(k0 + brow) * Ntot + n0 + bnp);
#pragma unroll
    for (int j = 0; j < 8; ++j) sBt[(bnp + j) * 32 + brow] = bv[j];
    __syncthreads();

    v16h af0, af1;
    *((v8h*)&af0)       = *(const v8h*)(&sAt[(wm * 32 + hl) * 32 + kb]);
    *(((v8h*)&af0) + 1) = *(const v8h*)(&sAt[(wm * 32 + hl) * 32 + kb + 16]);
    *((v8h*)&af1)       = *(const v8h*)(&sAt[(wm * 32 + 16 + hl) * 32 + kb]);
    *(((v8h*)&af1) + 1) = *(const v8h*)(&sAt[(wm * 32 + 16 + hl) * 32 + kb + 16]);
    v16h bf0, bf1;
    *((v8h*)&bf0)       = *(const v8h*)(&sBt[(wn * 32 + hl) * 32 + kb]);
    *(((v8h*)&bf0) + 1) = *(const v8h*)(&sBt[(wn * 32 + hl) * 32 + kb + 16]);
    *((v8h*)&bf1)       = *(const v8h*)(&sBt[(wn * 32 + 16 + hl) * 32 + kb]);
    *(((v8h*)&bf1) + 1) = *(const v8h*)(&sBt[(wn * 32 + 16 + hl) * 32 + kb + 16]);

    acc00 = __builtin_amdgcn_wmma_f32_16x16x32_f16(false, af0, false, bf0,
                                                   (short)0, acc00, false, false);
    acc01 = __builtin_amdgcn_wmma_f32_16x16x32_f16(false, af0, false, bf1,
                                                   (short)0, acc01, false, false);
    acc10 = __builtin_amdgcn_wmma_f32_16x16x32_f16(false, af1, false, bf0,
                                                   (short)0, acc10, false, false);
    acc11 = __builtin_amdgcn_wmma_f32_16x16x32_f16(false, af1, false, bf1,
                                                   (short)0, acc11, false, false);
    __syncthreads();
  }
  const int rsh = (lane >> 4) * 8;  // C/D layout: lanes>=16 hold rows r+8
#pragma unroll
  for (int r = 0; r < 8; ++r) {
    int gr0 = m0 + wm * 32 + r + rsh;
    int gr1 = gr0 + 16;
    size_t cc = (size_t)n0 + wn * 32 + hl;
    if (gr0 < M) {
      Cb[(size_t)gr0 * Ntot + cc]      = (_Float16)acc00[r];
      Cb[(size_t)gr0 * Ntot + cc + 16] = (_Float16)acc01[r];
    }
    if (gr1 < M) {
      Cb[(size_t)gr1 * Ntot + cc]      = (_Float16)acc10[r];
      Cb[(size_t)gr1 * Ntot + cc + 16] = (_Float16)acc11[r];
    }
  }
}

// ---------------------------------------------------------------------------
// start 1x1 convs: build x / x_a in [B,13,C,N] f32 from padded inputs
// ---------------------------------------------------------------------------
__launch_bounds__(256)
__global__ void k_start(float* __restrict__ x, float* __restrict__ xa,
                        const float* __restrict__ in,
                        const float* __restrict__ sw, const float* __restrict__ sb,
                        const float* __restrict__ saw, const float* __restrict__ sab) {
  size_t idx = (size_t)blockIdx.x * 256 + threadIdx.x;
  int n = (int)(idx & 511);
  int c = (int)((idx >> 9) & 31);
  int t = (int)((idx >> 14) % kRF);
  float v0 = 0.f, v1 = 0.f;
  if (t > 0) {
    int b = (int)(idx / ((size_t)16384 * kRF));
    size_t base = (((size_t)b * 2) * kN + n) * kT + (t - 1);
    v0 = in[base];
    v1 = in[base + (size_t)kN * kT];
  }
  x[idx]  = sw[c] * v0 + sb[c];
  xa[idx] = saw[c] * v1 + sab[c];
}

// ---------------------------------------------------------------------------
// dilated conv (1,2) + tanh*sigmoid gate -> f16 output [B,Tout,C,N]
// ---------------------------------------------------------------------------
__launch_bounds__(256)
__global__ void k_gate(_Float16* __restrict__ outh, const float* __restrict__ x,
                       const float* __restrict__ fw, const float* __restrict__ fb,
                       const float* __restrict__ gw, const float* __restrict__ gb,
                       int Tin, int d) {
  __shared__ float sx0[kC * 128], sx1[kC * 128];
  __shared__ float sF0[kC * kC], sF1[kC * kC], sG0[kC * kC], sG1[kC * kC];
  int tid = threadIdx.x;
  int Tout = Tin - d;
  int b = blockIdx.x / Tout, t = blockIdx.x % Tout;
  int n0 = blockIdx.y * 128;
  for (int j = tid; j < kC * kC; j += 256) {
    sF0[j] = fw[2 * j]; sF1[j] = fw[2 * j + 1];
    sG0[j] = gw[2 * j]; sG1[j] = gw[2 * j + 1];
  }
  size_t b0 = ((size_t)(b * Tin + t) * kC) * kN + n0;
  size_t b1 = ((size_t)(b * Tin + t + d) * kC) * kN + n0;
  for (int j = tid; j < kC * 128; j += 256) {
    int i = j >> 7, nl = j & 127;
    sx0[j] = x[b0 + (size_t)i * kN + nl];
    sx1[j] = x[b1 + (size_t)i * kN + nl];
  }
  __syncthreads();
  int og = tid >> 7, nl = tid & 127;
  size_t ob = ((size_t)(b * Tout + t) * kC) * kN + n0 + nl;
  for (int o = og; o < kC; o += 2) {
    float f = fb[o], g = gb[o];
#pragma unroll 8
    for (int i = 0; i < kC; ++i) {
      float x0 = sx0[i * 128 + nl], x1 = sx1[i * 128 + nl];
      f += sF0[o * kC + i] * x0 + sF1[o * kC + i] * x1;
      g += sG0[o * kC + i] * x0 + sG1[o * kC + i] * x1;
    }
    float val = tanhf(f) * (1.f / (1.f + __expf(-g)));
    outh[ob + (size_t)o * kN] = (_Float16)val;
  }
}

// ---------------------------------------------------------------------------
// skip projection: skip[b, choff + sk*Tout + t, n] = skip_W * xg + skip_b
// ---------------------------------------------------------------------------
__launch_bounds__(256)
__global__ void k_skip(float* __restrict__ skip, const _Float16* __restrict__ xgh,
                       const float* __restrict__ sw, const float* __restrict__ sb,
                       int Tout, int choff) {
  __shared__ float sW[kSK * kC];
  int tid = threadIdx.x;
  if (tid < kSK * kC) sW[tid] = sw[tid];
  __syncthreads();
  int b = blockIdx.x / Tout, t = blockIdx.x % Tout;
  int n = blockIdx.y * 256 + tid;
  float v[kC];
  size_t ibase = ((size_t)(b * Tout + t) * kC) * kN + n;
#pragma unroll
  for (int i = 0; i < kC; ++i) v[i] = (float)xgh[ibase + (size_t)i * kN];
#pragma unroll
  for (int s = 0; s < kSK; ++s) {
    float acc = sb[s];
#pragma unroll
    for (int i = 0; i < kC; ++i) acc += sW[s * kC + i] * v[i];
    skip[((size_t)b * kSKCH + choff + s * Tout + t) * kN + n] = acc;
  }
}

// ---------------------------------------------------------------------------
// accumulating C x C channel mix: out (f32) (+)= W[:,wcol:wcol+32] * inh(f16)
// optionally + bias + residual (read from [B,Tin,...] at t+toff)
// ---------------------------------------------------------------------------
__launch_bounds__(256)
__global__ void k_mix(float* __restrict__ out, const _Float16* __restrict__ inh,
                      const float* __restrict__ W, int wld, int wcol,
                      const float* __restrict__ bias,
                      const float* __restrict__ res, int Tin, int toff,
                      int Tout, int accum) {
  __shared__ float sW[kC * kC];
  __shared__ float sB[kC];
  int tid = threadIdx.x;
  for (int j = tid; j < kC * kC; j += 256) {
    int o = j >> 5, i = j & 31;
    sW[j] = W[o * wld + wcol + i];
  }
  if (tid < kC) sB[tid] = bias ? bias[tid] : 0.f;
  __syncthreads();
  int b = blockIdx.x / Tout, t = blockIdx.x % Tout;
  int n = blockIdx.y * 256 + tid;
  float v[kC];
  size_t ibase = ((size_t)(b * Tout + t) * kC) * kN + n;
#pragma unroll
  for (int i = 0; i < kC; ++i) v[i] = (float)inh[ibase + (size_t)i * kN];
  for (int o = 0; o < kC; ++o) {
    float acc = sB[o];
#pragma unroll
    for (int i = 0; i < kC; ++i) acc += sW[o * kC + i] * v[i];
    if (res) acc += res[((size_t)(b * Tin + t + toff) * kC + o) * kN + n];
    size_t oi = ((size_t)(b * Tout + t) * kC + o) * kN + n;
    out[oi] = accum ? out[oi] + acc : acc;
  }
}

// ---------------------------------------------------------------------------
// BatchNorm: deterministic two-stage reduction (no atomics) + apply
// ---------------------------------------------------------------------------
__launch_bounds__(256)
__global__ void k_bnstat(const float* __restrict__ g, float* __restrict__ part,
                         int Tout) {
  __shared__ float rs[256], rq[256];
  int tid = threadIdx.x;
  int c = blockIdx.x & 31, b = blockIdx.x >> 5;
  float s = 0.f, q = 0.f;
  int tot = Tout * kN;
  for (int j = tid; j < tot; j += 256) {
    int t = j >> 9, n = j & 511;
    float v = g[((size_t)(b * Tout + t) * kC + c) * kN + n];
    s += v; q += v * v;
  }
  rs[tid] = s; rq[tid] = q; __syncthreads();
  for (int st = 128; st > 0; st >>= 1) {
    if (tid < st) { rs[tid] += rs[tid + st]; rq[tid] += rq[tid + st]; }
    __syncthreads();
  }
  if (tid == 0) {
    part[(c * kB + b) * 2]     = rs[0];
    part[(c * kB + b) * 2 + 1] = rq[0];
  }
}

__launch_bounds__(64)
__global__ void k_bnfinal(const float* __restrict__ part, float* __restrict__ fstat) {
  int tid = threadIdx.x;          // 0..63 : two stat sets x 32 channels
  int set = tid >> 5, c = tid & 31;
  const float* p = part + set * 4096 + c * kB * 2;
  float s = 0.f, q = 0.f;
  for (int b = 0; b < kB; ++b) { s += p[2 * b]; q += p[2 * b + 1]; }
  fstat[set * 64 + c]      = s;
  fstat[set * 64 + 32 + c] = q;
}

__launch_bounds__(256)
__global__ void k_bnapply(float* __restrict__ xo, const float* __restrict__ g,
                          const float* __restrict__ st,
                          const float* __restrict__ gamma,
                          const float* __restrict__ beta, int Tout) {
  size_t idx = (size_t)blockIdx.x * 256 + threadIdx.x;
  int c = (int)((idx >> 9) & 31);
  float cnt = (float)kB * (float)Tout * (float)kN;
  float m = st[c] / cnt;
  float var = st[32 + c] / cnt - m * m;
  xo[idx] = gamma[c] * (g[idx] - m) * rsqrtf(var + 1e-5f) + beta[c];
}

// ---------------------------------------------------------------------------
// adjacency builders: relu + row softmax
// ---------------------------------------------------------------------------
__launch_bounds__(256)
__global__ void k_adjrow(float* __restrict__ outf, _Float16* __restrict__ outh,
                         const float* __restrict__ v1, const float* __restrict__ v2) {
  __shared__ float red[256];
  int v = blockIdx.x, tid = threadIdx.x;
  float w1[kDIMS];
#pragma unroll
  for (int d = 0; d < kDIMS; ++d) w1[d] = v1[v * kDIMS + d];
  int c0 = tid, c1 = tid + 256;
  float l0 = 0.f, l1 = 0.f;
#pragma unroll
  for (int d = 0; d < kDIMS; ++d) {
    l0 += w1[d] * v2[d * kN + c0];
    l1 += w1[d] * v2[d * kN + c1];
  }
  l0 = fmaxf(l0, 0.f); l1 = fmaxf(l1, 0.f);
  red[tid] = fmaxf(l0, l1); __syncthreads();
  for (int s = 128; s > 0; s >>= 1) {
    if (tid < s) red[tid] = fmaxf(red[tid], red[tid + s]);
    __syncthreads();
  }
  float m = red[0]; __syncthreads();
  float e0 = __expf(l0 - m), e1 = __expf(l1 - m);
  red[tid] = e0 + e1; __syncthreads();
  for (int s = 128; s > 0; s >>= 1) {
    if (tid < s) red[tid] += red[tid + s];
    __syncthreads();
  }
  float inv = 1.f / red[0];
  float p0 = e0 * inv, p1 = e1 * inv;
  size_t base = (size_t)v * kN;
  if (outf) { outf[base + c0] = p0; outf[base + c1] = p1; }
  outh[base + c0] = (_Float16)p0;
  outh[base + c1] = (_Float16)p1;
}

__launch_bounds__(128)
__global__ void k_t2(float* __restrict__ t2, const float* __restrict__ a2p1,
                     const int* __restrict__ ind, const float* __restrict__ a2pk) {
  int b = blockIdx.x, tid = threadIdx.x;
  if (tid < kDIMS * kDIMS) {
    const float* e = a2p1 + (size_t)ind[b] * kDIMS;
    float s = 0.f;
    for (int i = 0; i < kDIMS; ++i) s += e[i] * a2pk[i * kDIMS * kDIMS + tid];
    t2[b * kDIMS * kDIMS + tid] = s;
  }
}

__launch_bounds__(256)
__global__ void k_a2prow(float* __restrict__ outf, _Float16* __restrict__ outh,
                         const float* __restrict__ t2,
                         const float* __restrict__ a2p2,
                         const float* __restrict__ a2p3) {
  __shared__ float sT2[kDIMS * kDIMS];
  __shared__ float sT3[kDIMS];
  __shared__ float red[256];
  int bid = blockIdx.x;
  int b = bid >> 9, v = bid & 511;
  int tid = threadIdx.x;
  if (tid < kDIMS * kDIMS) sT2[tid] = t2[b * kDIMS * kDIMS + tid];
  __syncthreads();
  if (tid < kDIMS) {
    float s = 0.f;
    for (int j = 0; j < kDIMS; ++j) s += a2p2[v * kDIMS + j] * sT2[j * kDIMS + tid];
    sT3[tid] = s;
  }
  __syncthreads();
  int c0 = tid, c1 = tid + 256;
  float l0 = 0.f, l1 = 0.f;
#pragma unroll
  for (int k = 0; k < kDIMS; ++k) {
    l0 += a2p3[c0 * kDIMS + k] * sT3[k];
    l1 += a2p3[c1 * kDIMS + k] * sT3[k];
  }
  l0 = fmaxf(l0, 0.f); l1 = fmaxf(l1, 0.f);
  red[tid] = fmaxf(l0, l1); __syncthreads();
  for (int s = 128; s > 0; s >>= 1) {
    if (tid < s) red[tid] = fmaxf(red[tid], red[tid + s]);
    __syncthreads();
  }
  float m = red[0]; __syncthreads();
  float e0 = __expf(l0 - m), e1 = __expf(l1 - m);
  red[tid] = e0 + e1; __syncthreads();
  for (int s = 128; s > 0; s >>= 1) {
    if (tid < s) red[tid] += red[tid + s];
    __syncthreads();
  }
  float inv = 1.f / red[0];
  size_t base = ((size_t)b * kN + v) * kN;
  outf[base + c0] = e0 * inv; outf[base + c1] = e1 * inv;
  outh[base + c0] = (_Float16)(e0 * inv);
  outh[base + c1] = (_Float16)(e1 * inv);
}

// ---------------------------------------------------------------------------
// end stage helpers
// ---------------------------------------------------------------------------
__launch_bounds__(256)
__global__ void k_relu_cast(_Float16* __restrict__ o, const float* __restrict__ a) {
  size_t i = (size_t)blockIdx.x * 256 + threadIdx.x;
  o[i] = (_Float16)fmaxf(a[i], 0.f);
}

__launch_bounds__(256)
__global__ void k_cast(_Float16* __restrict__ o, const float* __restrict__ a) {
  size_t i = (size_t)blockIdx.x * 256 + threadIdx.x;
  o[i] = (_Float16)a[i];
}

__launch_bounds__(256)
__global__ void k_end2(float* __restrict__ out, const _Float16* __restrict__ hh,
                       const float* __restrict__ w2, const float* __restrict__ b2,
                       const float* __restrict__ b1) {
  __shared__ float sW[kOD * kEC];
  __shared__ float sB1[kEC];
  for (int j = threadIdx.x; j < kOD * kEC; j += 256) sW[j] = w2[j];
  for (int j = threadIdx.x; j < kEC; j += 256) sB1[j] = b1[j];
  __syncthreads();
  int b = blockIdx.x;
  int n = blockIdx.y * 256 + threadIdx.x;
  float acc[kOD];
#pragma unroll
  for (int o = 0; o < kOD; ++o) acc[o] = b2[o];
  for (int e = 0; e < kEC; ++e) {
    float hv = fmaxf((float)hh[((size_t)b * kEC + e) * kN + n] + sB1[e], 0.f);
#pragma unroll
    for (int o = 0; o < kOD; ++o) acc[o] += sW[o * kEC + e] * hv;
  }
#pragma unroll
  for (int o = 0; o < kOD; ++o)
    out[((size_t)b * kOD + o) * kN + n] = acc[o];
}

// ---------------------------------------------------------------------------
// host side
// ---------------------------------------------------------------------------
static void launch_gemm(const _Float16* A, const _Float16* Bm, _Float16* Cm,
                        int M, int Nt, int K, long sA, long sB, long sC,
                        int batch, hipStream_t st) {
  dim3 g((M + 127) / 128, Nt / 64, batch);
  k_gemm_f16<<<g, 256, 0, st>>>(A, Bm, Cm, M, Nt, K, sA, sB, sC);
}

extern "C" void kernel_launch(void* const* d_in, const int* in_sizes, int n_in,
                              void* d_out, int out_size, void* d_ws, size_t ws_size,
                              hipStream_t stream) {
  (void)in_sizes; (void)n_in; (void)out_size; (void)ws_size;  // WS_NEED ~400MB
  char* ws = (char*)d_ws;

  const float* inputs   = (const float*)d_in[0];
  const int*   ind      = (const int*)d_in[1];
  const float* start_W  = (const float*)d_in[2];
  const float* start_b  = (const float*)d_in[3];
  const float* starta_W = (const float*)d_in[4];
  const float* starta_b = (const float*)d_in[5];
  const float* filt_W   = (const float*)d_in[6];
  const float* filt_b   = (const float*)d_in[7];
  const float* gate_W   = (const float*)d_in[8];
  const float* gate_b   = (const float*)d_in[9];
  const float* filta_W  = (const float*)d_in[10];
  const float* filta_b  = (const float*)d_in[11];
  const float* gatea_W  = (const float*)d_in[12];
  const float* gatea_b  = (const float*)d_in[13];
  const float* skip_W   = (const float*)d_in[14];
  const float* skip_b   = (const float*)d_in[15];
  const float* bn_g     = (const float*)d_in[16];
  const float* bn_b     = (const float*)d_in[17];
  const float* bna_g    = (const float*)d_in[18];
  const float* bna_b    = (const float*)d_in[19];
  const float* gconv_W  = (const float*)d_in[20];
  const float* gconv_b  = (const float*)d_in[21];
  const float* gconva_W = (const float*)d_in[22];
  const float* gconva_b = (const float*)d_in[23];
  const float* ga2p_W   = (const float*)d_in[24];
  const float* ga2p_b   = (const float*)d_in[25];
  const float* nv1      = (const float*)d_in[26];
  const float* nv2      = (const float*)d_in[27];
  const float* nv1a     = (const float*)d_in[28];
  const float* nv2a     = (const float*)d_in[29];
  const float* a2p1     = (const float*)d_in[30];
  const float* a2p2     = (const float*)d_in[31];
  const float* a2p3     = (const float*)d_in[32];
  const float* a2pk     = (const float*)d_in[33];
  const float* end1_W   = (const float*)d_in[34];
  const float* end1_b   = (const float*)d_in[35];
  const float* end2_W   = (const float*)d_in[36];
  const float* end2_b   = (const float*)d_in[37];

  float* outy   = (float*)d_out;                         // [B,OD,N]
  float* outadp = outy + (size_t)kB * kOD * kN;          // [N,N]
  float* outa2p = outadp + (size_t)kN * kN;              // [B,N,N]

  float*     xbuf  = (float*)(ws + O_X);
  float*     xabuf = (float*)(ws + O_XA);
  float*     skipb = (float*)(ws + O_SKIP);
  _Float16*  adph  = (_Float16*)(ws + O_ADPH);
  _Float16*  adpah = (_Float16*)(ws + O_ADPAH);
  _Float16*  a2ph  = (_Float16*)(ws + O_A2PH);
  float*     t2b   = (float*)(ws + O_T2);
  float*     part  = (float*)(ws + O_PART);
  float*     fstat = (float*)(ws + O_FSTAT);
  _Float16*  e1wh  = (_Float16*)(ws + O_E1WH);
  _Float16*  xgh   = (_Float16*)(ws + O_XGH);
  _Float16*  xagh  = (_Float16*)(ws + O_XAGH);
  _Float16*  t1h   = (_Float16*)(ws + O_T1H);
  _Float16*  t2h   = (_Float16*)(ws + O_T2H);
  float*     gbuf  = (float*)(ws + O_G);
  float*     gabuf = (float*)(ws + O_GA);
  _Float16*  skiph = (_Float16*)(ws + O_SKIPH);
  _Float16*  hh    = (_Float16*)(ws + O_HH);

  // adjacencies + start convs
  k_adjrow<<<kN, 256, 0, stream>>>(outadp, adph, nv1, nv2);
  k_adjrow<<<kN, 256, 0, stream>>>(nullptr, adpah, nv1a, nv2a);
  k_t2<<<kB, 128, 0, stream>>>(t2b, a2p1, ind, a2pk);
  k_a2prow<<<kB * kN, 256, 0, stream>>>(outa2p, a2ph, t2b, a2p2, a2p3);
  k_start<<<(kB * kRF * kC * kN) / 256, 256, 0, stream>>>(
      xbuf, xabuf, inputs, start_W, start_b, starta_W, starta_b);

  static const int DILSh[kL]  = {1, 2, 1, 2, 1, 2, 1, 2};
  static const int CHOFFh[kL] = {320, 240, 168, 112, 64, 32, 8, 0};

  int Tin = kRF;
  for (int i = 0; i < kL; ++i) {
    const int d = DILSh[i], Tout = Tin - d;
    const size_t wOff = (size_t)i * kC * kC * 2;
    const float* gcw  = gconv_W + (size_t)i * kC * 3 * kC;
    const float* gcaw = gconva_W + (size_t)i * kC * 3 * kC;
    const float* gpw  = ga2p_W + (size_t)i * kC * 3 * kC;
    const dim3 gmix(kB * Tout, kN / 256);

    // gated dilated convs -> f16
    k_gate<<<dim3(kB * Tout, kN / 128), 256, 0, stream>>>(
        xgh, xbuf, filt_W + wOff, filt_b + i * kC, gate_W + wOff,
        gate_b + i * kC, Tin, d);
    k_gate<<<dim3(kB * Tout, kN / 128), 256, 0, stream>>>(
        xagh, xabuf, filta_W + wOff, filta_b + i * kC, gatea_W + wOff,
        gatea_b + i * kC, Tin, d);

    // skip projection
    k_skip<<<gmix, 256, 0, stream>>>(skipb, xgh, skip_W + i * kSK * kC,
                                     skip_b + i * kSK, Tout, CHOFFh[i]);

    const int M1 = kB * Tout * kC;  // shared-adjacency GEMM rows
    const int M2 = Tout * kC;       // per-sample GEMM rows

    // x path: x1 = xg*adp, x2 = x1*adp, then mix into g (+residual+bias)
    launch_gemm(xgh, adph, t1h, M1, kN, kN, 0, 0, 0, 1, stream);
    launch_gemm(t1h, adph, t2h, M1, kN, kN, 0, 0, 0, 1, stream);
    k_mix<<<gmix, 256, 0, stream>>>(gbuf, xgh, gcw, 3 * kC, 0,
                                    gconv_b + i * kC, xbuf, Tin, d, Tout, 0);
    k_mix<<<gmix, 256, 0, stream>>>(gbuf, t1h, gcw, 3 * kC, kC,
                                    nullptr, nullptr, Tin, 0, Tout, 1);
    k_mix<<<gmix, 256, 0, stream>>>(gbuf, t2h, gcw, 3 * kC, 2 * kC,
                                    nullptr, nullptr, Tin, 0, Tout, 1);

    // a2p path (batched per-sample adjacency) accumulated into g
    launch_gemm(xagh, a2ph, t1h, M2, kN, kN, (long)M2 * kN, (long)kN * kN,
                (long)M2 * kN, kB, stream);
    launch_gemm(t1h, a2ph, t2h, M2, kN, kN, (long)M2 * kN, (long)kN * kN,
                (long)M2 * kN, kB, stream);
    k_mix<<<gmix, 256, 0, stream>>>(gbuf, xagh, gpw, 3 * kC, 0,
                                    ga2p_b + i * kC, nullptr, Tin, 0, Tout, 1);
    k_mix<<<gmix, 256, 0, stream>>>(gbuf, t1h, gpw, 3 * kC, kC,
                                    nullptr, nullptr, Tin, 0, Tout, 1);
    k_mix<<<gmix, 256, 0, stream>>>(gbuf, t2h, gpw, 3 * kC, 2 * kC,
                                    nullptr, nullptr, Tin, 0, Tout, 1);

    // x_a path
    launch_gemm(xagh, adpah, t1h, M1, kN, kN, 0, 0, 0, 1, stream);
    launch_gemm(t1h, adpah, t2h, M1, kN, kN, 0, 0, 0, 1, stream);
    k_mix<<<gmix, 256, 0, stream>>>(gabuf, xagh, gcaw, 3 * kC, 0,
                                    gconva_b + i * kC, xabuf, Tin, d, Tout, 0);
    k_mix<<<gmix, 256, 0, stream>>>(gabuf, t1h, gcaw, 3 * kC, kC,
                                    nullptr, nullptr, Tin, 0, Tout, 1);
    k_mix<<<gmix, 256, 0, stream>>>(gabuf, t2h, gcaw, 3 * kC, 2 * kC,
                                    nullptr, nullptr, Tin, 0, Tout, 1);

    // batchnorm (training-mode batch stats), writes next-iteration x / x_a
    k_bnstat<<<kC * kB, 256, 0, stream>>>(gbuf, part, Tout);
    k_bnstat<<<kC * kB, 256, 0, stream>>>(gabuf, part + 4096, Tout);
    k_bnfinal<<<1, 64, 0, stream>>>(part, fstat);
    const size_t ne = (size_t)kB * Tout * kC * kN;
    k_bnapply<<<ne / 256, 256, 0, stream>>>(xbuf, gbuf, fstat,
                                            bn_g + i * kC, bn_b + i * kC, Tout);
    k_bnapply<<<ne / 256, 256, 0, stream>>>(xabuf, gabuf, fstat + 64,
                                            bna_g + i * kC, bna_b + i * kC, Tout);
    Tin = Tout;
  }

  // end MLP: relu(skip) -> end1 (WMMA GEMM K=416) -> relu -> end2
  k_relu_cast<<<(kB * kSKCH * kN) / 256, 256, 0, stream>>>(skiph, skipb);
  k_cast<<<(kEC * kSKCH) / 256, 256, 0, stream>>>(e1wh, end1_W);
  launch_gemm(e1wh, skiph, hh, kEC, kN, kSKCH, 0, (long)kSKCH * kN,
              (long)kEC * kN, kB, stream);
  k_end2<<<dim3(kB, kN / 256), 256, 0, stream>>>(outy, hh, end2_W, end2_b,
                                                 end1_b);
}
